// Attention_88184268521490
// MI455X (gfx1250) — compile-verified, matrix operand order checked by reading, not run on hardware
//
#include <hip/hip_runtime.h>
#include <math.h>

typedef __attribute__((ext_vector_type(16))) _Float16 v16h;
typedef __attribute__((ext_vector_type(8)))  _Float16 v8h;
typedef __attribute__((ext_vector_type(8)))  float    v8f;

#define NH 8
#define CH 32
#define CQ 256
#define BB 2
#define QL 2048
#define KL 2048

static __device__ __forceinline__ v16h cat8(v8h a, v8h b) {
  return __builtin_shufflevector(a, b, 0,1,2,3,4,5,6,7,8,9,10,11,12,13,14,15);
}
static __device__ __forceinline__ v16h cvt2(v8f a, v8f b) {
  v16h r;
#pragma unroll
  for (int i = 0; i < 8; ++i) { r[i] = (_Float16)a[i]; r[8 + i] = (_Float16)b[i]; }
  return r;
}
static __device__ __forceinline__ v8f wmma16(v16h a, v16h b, v8f c) {
  return __builtin_amdgcn_wmma_f32_16x16x32_f16(false, a, false, b, (short)0, c, false, false);
}

// ---- prep: transpose 256x256 f32 weight -> f16 row-major of W^T ----
__global__ void k_transpose(const float* __restrict__ w, _Float16* __restrict__ wt) {
  int i = threadIdx.x;   // input row (k)
  int j = blockIdx.x;    // input col (n) -> output row
  wt[(size_t)j * CQ + i] = (_Float16)w[(size_t)i * CQ + j];
}

// ---- projections: [B*L,256] @ [256,256], one wave per 16x16 tile ----
// mode 0: q (scale 1/sqrt(32), split-head [b,h,l,32] f16)
// mode 1: k (split-head f16)
// mode 2: v (transposed [b,h,c,L] f16)
// mode 3: gate (sigmoid(x+b_g), f32 [B*L,256])
__global__ void k_proj(const float* __restrict__ x, const _Float16* __restrict__ wt,
                       _Float16* __restrict__ out_h, float* __restrict__ out_g,
                       const float* __restrict__ bias, int mode, int L) {
  const int lane = threadIdx.x;
  const int lo = lane & 15, hh = lane >> 4;
  const int m0 = blockIdx.x * 16;
  const int n0 = blockIdx.y * 16;
  const float*    xr = x  + (size_t)(m0 + lo) * CQ;
  const _Float16* wr = wt + (size_t)(n0 + lo) * CQ;
  const int n = n0 + lo;
  v8f acc = {};
#pragma unroll
  for (int kc = 0; kc < CQ; kc += 32) {
    v8f a0 = *(const v8f*)(xr + kc + hh * 8);
    v8f a1 = *(const v8f*)(xr + kc + 16 + hh * 8);
    v16h a  = cvt2(a0, a1);
    v16h bm = cat8(*(const v8h*)(wr + kc + hh * 8),
                   *(const v8h*)(wr + kc + 16 + hh * 8));
    acc = wmma16(a, bm, acc);
  }
#pragma unroll
  for (int r = 0; r < 8; ++r) {
    int mm = m0 + 8 * hh + r;           // global row in [0, B*L)
    int b = mm / L, li = mm % L;
    int h = n / CH, c = n % CH;
    float v = acc[r];
    if (mode == 0) {
      out_h[(((size_t)(b * NH + h) * L) + li) * CH + c] =
          (_Float16)(v * 0.17677669529663687f);
    } else if (mode == 1) {
      out_h[(((size_t)(b * NH + h) * L) + li) * CH + c] = (_Float16)v;
    } else if (mode == 2) {
      out_h[(((size_t)(b * NH + h) * CH) + c) * (size_t)L + li] = (_Float16)v;
    } else {
      out_g[(size_t)mm * CQ + n] = 1.0f / (1.0f + __expf(-(v + bias[n])));
    }
  }
}

// ---- fused flash attention, one wave per (b, h, 16 q rows) ----
// K/V tiles are double-buffered in LDS via global_load_async_to_lds_b128
// (ASYNCcnt-tracked), so the next 32-key tile streams in while the current
// tile's WMMAs + online softmax execute.
__global__ void k_attn(const _Float16* __restrict__ qh, const _Float16* __restrict__ kh,
                       const _Float16* __restrict__ vt, const float* __restrict__ bias_mask,
                       const float* __restrict__ bias_pair, const float* __restrict__ g,
                       _Float16* __restrict__ og) {
  __shared__ __align__(16) _Float16 kbuf[2][32][32];
  __shared__ __align__(16) _Float16 vbuf[2][32][32];
  __shared__ __align__(16) _Float16 lds_p[16][32];
  const int lane = threadIdx.x;
  const int lo = lane & 15, hh = lane >> 4;
  const int q0 = blockIdx.x * 16;
  const int h  = blockIdx.y;
  const int b  = blockIdx.z;
  const size_t bh = (size_t)(b * NH + h);

  // async-copy a 32x32 f16 tile (rows have stride rowstride halves) into LDS
  auto async_tile = [&](const _Float16* gbase, size_t rowstride, _Float16 (*dst)[32]) {
#pragma unroll
    for (int j = 0; j < 4; ++j) {
      int chunk = lane + 32 * j;        // 0..127 sixteen-byte chunks
      int row  = chunk >> 2;
      int c16  = (chunk & 3) * 8;       // halves
      const _Float16* gp = gbase + (size_t)row * rowstride + c16;
      unsigned ldsoff = (unsigned)(uintptr_t)(&dst[row][c16]);
      asm volatile("global_load_async_to_lds_b128 %0, %1, off"
                   :: "v"(ldsoff), "v"(gp) : "memory");
    }
  };

  // Q tile as WMMA A operand (K-dim = 32 channels, loaded once)
  const _Float16* qr = qh + (bh * QL + q0 + lo) * CH;
  v16h aq = cat8(*(const v8h*)(qr + hh * 8), *(const v8h*)(qr + 16 + hh * 8));

  // prefetch tile 0
  async_tile(kh + (bh * KL) * CH, CH, kbuf[0]);
  async_tile(vt + (bh * CH) * (size_t)KL, KL, vbuf[0]);

  float mrow[8], lrow[8];
#pragma unroll
  for (int r = 0; r < 8; ++r) { mrow[r] = -1e30f; lrow[r] = 0.0f; }
  v8f o0 = {}, o1 = {};

  for (int kc = 0; kc < KL; kc += 32) {
    const int buf = (kc >> 5) & 1;
    if (kc + 32 < KL) {
      // issue next tile into the other buffer, then wait for current tile:
      // the 8 newest async ops may stay outstanding (async loads complete
      // in order), so <=8 outstanding means the current buffer is ready.
      async_tile(kh + (bh * KL + kc + 32) * CH, CH, kbuf[buf ^ 1]);
      async_tile(vt + (bh * CH) * (size_t)KL + (kc + 32), KL, vbuf[buf ^ 1]);
      asm volatile("s_wait_asynccnt 0x8" ::: "memory");
    } else {
      asm volatile("s_wait_asynccnt 0x0" ::: "memory");
    }

    // B operands: rows of K (== columns of K^T) from LDS
    v16h b0 = cat8(*(const v8h*)(&kbuf[buf][lo][hh * 8]),
                   *(const v8h*)(&kbuf[buf][lo][16 + hh * 8]));
    v16h b1 = cat8(*(const v8h*)(&kbuf[buf][16 + lo][hh * 8]),
                   *(const v8h*)(&kbuf[buf][16 + lo][16 + hh * 8]));
    v8f z = {};
    v8f s0 = wmma16(aq, b0, z);   // S columns kc..kc+15
    v8f s1 = wmma16(aq, b1, z);   // S columns kc+16..kc+31

    float bm0 = bias_mask[(size_t)b * KL + kc + lo];
    float bm1 = bias_mask[(size_t)b * KL + kc + 16 + lo];
    float p0[8], p1[8], alpha[8], cm[8];
#pragma unroll
    for (int r = 0; r < 8; ++r) {
      int qq = q0 + 8 * hh + r;
      const float* bp = bias_pair + ((size_t)h * QL + qq) * KL + kc;
      s0[r] += bm0 + bp[lo];
      s1[r] += bm1 + bp[16 + lo];
      cm[r] = fmaxf(s0[r], s1[r]);
    }
    if (kc + 32 < KL) {  // prefetch next bias_pair chunk
      const float* bp = bias_pair + ((size_t)h * QL + q0 + 8 * hh) * KL + kc + 32;
      __builtin_prefetch(bp + lane, 0, 0);
    }
    // row max across the 16 columns held by this 16-lane group
#pragma unroll
    for (int s = 1; s <= 8; s <<= 1)
#pragma unroll
      for (int r = 0; r < 8; ++r)
        cm[r] = fmaxf(cm[r], __shfl_xor(cm[r], s, 32));
    float rs[8];
#pragma unroll
    for (int r = 0; r < 8; ++r) {
      float mn = fmaxf(mrow[r], cm[r]);
      alpha[r] = __expf(mrow[r] - mn);
      mrow[r] = mn;
      p0[r] = __expf(s0[r] - mn);
      p1[r] = __expf(s1[r] - mn);
      rs[r] = p0[r] + p1[r];
    }
#pragma unroll
    for (int s = 1; s <= 8; s <<= 1)
#pragma unroll
      for (int r = 0; r < 8; ++r)
        rs[r] += __shfl_xor(rs[r], s, 32);
#pragma unroll
    for (int r = 0; r < 8; ++r) {
      lrow[r] = lrow[r] * alpha[r] + rs[r];
      o0[r] *= alpha[r];
      o1[r] *= alpha[r];
    }
    // transpose P from D layout to A layout via LDS
    __syncthreads();
#pragma unroll
    for (int r = 0; r < 8; ++r) {
      lds_p[8 * hh + r][lo]      = (_Float16)p0[r];
      lds_p[8 * hh + r][16 + lo] = (_Float16)p1[r];
    }
    __syncthreads();
    v16h pa = cat8(*(const v8h*)(&lds_p[lo][hh * 8]),
                   *(const v8h*)(&lds_p[lo][16 + hh * 8]));
    // B operands: rows of V^T from LDS, output channels lo and 16+lo
    v16h bv0 = cat8(*(const v8h*)(&vbuf[buf][lo][hh * 8]),
                    *(const v8h*)(&vbuf[buf][lo][16 + hh * 8]));
    v16h bv1 = cat8(*(const v8h*)(&vbuf[buf][16 + lo][hh * 8]),
                    *(const v8h*)(&vbuf[buf][16 + lo][16 + hh * 8]));
    o0 = wmma16(pa, bv0, o0);
    o1 = wmma16(pa, bv1, o1);
  }
  // epilogue: normalize, gate, store f16 [B*Q, H*CH]
#pragma unroll
  for (int r = 0; r < 8; ++r) {
    int qq = q0 + 8 * hh + r;
    float inv = 1.0f / lrow[r];
    size_t base = ((size_t)b * QL + qq) * (NH * CH);
    int j0 = h * CH + lo;
    int j1 = h * CH + 16 + lo;
    og[base + j0] = (_Float16)(o0[r] * inv * g[base + j0]);
    og[base + j1] = (_Float16)(o1[r] * inv * g[base + j1]);
  }
}

// ---- final GEMM: [B*Q,256] @ w_o + b_o -> f32 out ----
__global__ void k_out(const _Float16* __restrict__ og, const _Float16* __restrict__ wot,
                      const float* __restrict__ b_o, float* __restrict__ out) {
  const int lane = threadIdx.x;
  const int lo = lane & 15, hh = lane >> 4;
  const int m0 = blockIdx.x * 16, n0 = blockIdx.y * 16;
  const _Float16* ar = og  + (size_t)(m0 + lo) * CQ;
  const _Float16* wr = wot + (size_t)(n0 + lo) * CQ;
  const int n = n0 + lo;
  v8f acc = {};
#pragma unroll
  for (int kc = 0; kc < CQ; kc += 32) {
    v16h a  = cat8(*(const v8h*)(ar + kc + hh * 8), *(const v8h*)(ar + kc + 16 + hh * 8));
    v16h bm = cat8(*(const v8h*)(wr + kc + hh * 8), *(const v8h*)(wr + kc + 16 + hh * 8));
    acc = wmma16(a, bm, acc);
  }
#pragma unroll
  for (int r = 0; r < 8; ++r) {
    int mm = m0 + 8 * hh + r;
    out[(size_t)mm * CQ + n] = acc[r] + b_o[n];
  }
}

extern "C" void kernel_launch(void* const* d_in, const int* in_sizes, int n_in,
                              void* d_out, int out_size, void* d_ws, size_t ws_size,
                              hipStream_t stream) {
  const float* q_x       = (const float*)d_in[0];
  const float* kv_x      = (const float*)d_in[1];
  const float* bias_mask = (const float*)d_in[2];
  const float* bias_pair = (const float*)d_in[3];
  const float* w_q       = (const float*)d_in[4];
  const float* w_k       = (const float*)d_in[5];
  const float* w_v       = (const float*)d_in[6];
  const float* w_g       = (const float*)d_in[7];
  const float* b_g       = (const float*)d_in[8];
  const float* w_o       = (const float*)d_in[9];
  const float* b_o       = (const float*)d_in[10];
  float* out = (float*)d_out;

  char* ws = (char*)d_ws;
  size_t off = 0;
  auto alloc = [&](size_t bytes) {
    size_t o = off;
    off += (bytes + 255) & ~(size_t)255;
    return o;
  };
  _Float16* wqt = (_Float16*)(ws + alloc((size_t)CQ * CQ * 2));
  _Float16* wkt = (_Float16*)(ws + alloc((size_t)CQ * CQ * 2));
  _Float16* wvt = (_Float16*)(ws + alloc((size_t)CQ * CQ * 2));
  _Float16* wgt = (_Float16*)(ws + alloc((size_t)CQ * CQ * 2));
  _Float16* wot = (_Float16*)(ws + alloc((size_t)CQ * CQ * 2));
  _Float16* qh  = (_Float16*)(ws + alloc((size_t)BB * NH * QL * CH * 2));
  _Float16* kh  = (_Float16*)(ws + alloc((size_t)BB * NH * KL * CH * 2));
  _Float16* vtp = (_Float16*)(ws + alloc((size_t)BB * NH * CH * KL * 2));
  float*    gb  = (float*)   (ws + alloc((size_t)BB * QL * CQ * 4));
  _Float16* og  = (_Float16*)(ws + alloc((size_t)BB * QL * CQ * 2));

  // weight transposes (f32 -> f16 W^T)
  k_transpose<<<dim3(CQ), dim3(CQ), 0, stream>>>(w_q, wqt);
  k_transpose<<<dim3(CQ), dim3(CQ), 0, stream>>>(w_k, wkt);
  k_transpose<<<dim3(CQ), dim3(CQ), 0, stream>>>(w_v, wvt);
  k_transpose<<<dim3(CQ), dim3(CQ), 0, stream>>>(w_g, wgt);
  k_transpose<<<dim3(CQ), dim3(CQ), 0, stream>>>(w_o, wot);

  // projections: one wave (32 threads) per 16x16 tile
  dim3 pgrid((BB * QL) / 16, CQ / 16);
  k_proj<<<pgrid, 32, 0, stream>>>(q_x,  wqt, qh,  nullptr, nullptr, 0, QL);
  k_proj<<<pgrid, 32, 0, stream>>>(kv_x, wkt, kh,  nullptr, nullptr, 1, KL);
  k_proj<<<pgrid, 32, 0, stream>>>(kv_x, wvt, vtp, nullptr, nullptr, 2, KL);
  k_proj<<<pgrid, 32, 0, stream>>>(q_x,  wgt, nullptr, gb,  b_g,     3, QL);

  // fused flash attention + gating (async double-buffered K/V tiles)
  k_attn<<<dim3(QL / 16, NH, BB), 32, 0, stream>>>(qh, kh, vtp, bias_mask,
                                                   bias_pair, gb, og);

  // output projection
  k_out<<<dim3((BB * QL) / 16, CQ / 16), 32, 0, stream>>>(og, wot, b_o, out);
}
